// RelMultiHeadAttention_29807073034411
// MI455X (gfx1250) — compile-verified, matrix operand order checked by reading, not run on hardware
//
#include <hip/hip_runtime.h>

typedef __attribute__((ext_vector_type(16))) __bf16 bf16x16;
typedef __attribute__((ext_vector_type(8)))  __bf16 bf16x8;
typedef __attribute__((ext_vector_type(8)))  float  f32x8;
typedef __attribute__((ext_vector_type(4)))  int    i32x4;

__device__ __forceinline__ f32x8 wmma_bf16(bf16x16 a, bf16x16 b, f32x8 c) {
  return __builtin_amdgcn_wmma_f32_16x16x32_bf16(false, a, false, b, (short)0, c,
                                                 false, false);
}

// Load one 16x32 A-frag / 32x16 B-frag lane slice: elements [p..p+7] and [p+16..p+23].
__device__ __forceinline__ bf16x16 frag16(const __bf16* p) {
  bf16x8 lo = *(const bf16x8*)p;
  bf16x8 hi = *(const bf16x8*)(p + 16);
  return __builtin_shufflevector(lo, hi, 0, 1, 2, 3, 4, 5, 6, 7,
                                 8, 9, 10, 11, 12, 13, 14, 15);
}

// ---------------- gfx1250 async global->LDS copy (16B per lane), guarded ----------------
#if __has_builtin(__builtin_amdgcn_global_load_async_to_lds_b128)
#define HAS_ASYNC_LDS 1
#else
#define HAS_ASYNC_LDS 0
#endif

__device__ __forceinline__ void async_copy16(const __bf16* g, __bf16* l) {
#if HAS_ASYNC_LDS
  __builtin_amdgcn_global_load_async_to_lds_b128(
      (__attribute__((address_space(1))) i32x4*)g,
      (__attribute__((address_space(3))) i32x4*)l, 0, 0);
#else
  *(bf16x8*)l = *(const bf16x8*)g;
#endif
}

__device__ __forceinline__ void wait_async0() {
#if HAS_ASYNC_LDS
#if __has_builtin(__builtin_amdgcn_s_wait_asynccnt)
  __builtin_amdgcn_s_wait_asynccnt(0);
#else
  asm volatile("s_wait_asynccnt 0x0" ::: "memory");
#endif
#endif
}

// ---------------------------------------------------------------- conversions (x8 vectorized)
__global__ void cvt_f32_bf16(const float* __restrict__ s, __bf16* __restrict__ d, int n) {
  int i = (blockIdx.x * 256 + threadIdx.x) * 8;
  if (i < n) {
    float4 a = *(const float4*)(s + i);
    float4 b = *(const float4*)(s + i + 4);
    bf16x8 o;
    o[0] = (__bf16)a.x; o[1] = (__bf16)a.y; o[2] = (__bf16)a.z; o[3] = (__bf16)a.w;
    o[4] = (__bf16)b.x; o[5] = (__bf16)b.y; o[6] = (__bf16)b.z; o[7] = (__bf16)b.w;
    *(bf16x8*)(d + i) = o;
  }
}

// cat = concat(mems, inputs) along time -> bf16 (B=4, T=2048, D=1024)
__global__ void build_cat(const float* __restrict__ mems, const float* __restrict__ inputs,
                          __bf16* __restrict__ cat) {
  size_t i = ((size_t)blockIdx.x * 256 + threadIdx.x) * 8;   // < 4*2048*1024
  int d = (int)(i & 1023);
  size_t bt = i >> 10;
  int t = (int)(bt & 2047);
  int b = (int)(bt >> 11);
  const float* src = (t < 1024)
                         ? mems + ((((size_t)b * 1024) + t) * 1024 + d)
                         : inputs + ((((size_t)b * 1024) + (t - 1024)) * 1024 + d);
  float4 a = *(const float4*)src;
  float4 c = *(const float4*)(src + 4);
  bf16x8 o;
  o[0] = (__bf16)a.x; o[1] = (__bf16)a.y; o[2] = (__bf16)a.z; o[3] = (__bf16)a.w;
  o[4] = (__bf16)c.x; o[5] = (__bf16)c.y; o[6] = (__bf16)c.z; o[7] = (__bf16)c.w;
  *(bf16x8*)(cat + i) = o;
}

// ---------------------------------------------------------------- GEMM (bf16 x bf16 -> f32)
// C(MxN) = A(MxK) @ B(KxN), tiles 128x128x32, 256 threads = 8 waves, wave owns 16xN strip.
// Double-buffered: A via async global->LDS, B via regs (transpose scatter after compute).
// EPI 0: QKV scatter (+biases), EPI 1: bf16 out, EPI 2: f32 out.
template <int EPI>
__global__ __launch_bounds__(256) void gemm_bf16_kernel(
    const __bf16* __restrict__ A, const __bf16* __restrict__ B,
    int M, int N, int K,
    __bf16* __restrict__ o0, __bf16* __restrict__ o1,
    __bf16* __restrict__ o2, __bf16* __restrict__ o3,
    float* __restrict__ of,
    const float* __restrict__ bias0, const float* __restrict__ bias1) {
  __shared__ __align__(16) __bf16 As[2][128][40];   // row-major 128x32, padded
  __shared__ __align__(16) __bf16 Bs[2][128][40];   // B^T: [n][k], padded

  const int tid = threadIdx.x;
  const int w = tid >> 5, lane = tid & 31;
  const int hs = lane >> 4, ln = lane & 15, kb = hs * 8;
  const int m0 = blockIdx.y * 128, n0 = blockIdx.x * 128;

  f32x8 zero = {0.f, 0.f, 0.f, 0.f, 0.f, 0.f, 0.f, 0.f};
  f32x8 acc[8] = {zero, zero, zero, zero, zero, zero, zero, zero};

  const int arow = tid >> 1, aseg = (tid & 1) * 16;   // A: 128 rows x 32
  const int brow = tid >> 3, bseg = (tid & 7) * 16;   // B: 32 rows x 128

  const __bf16* arp = A + (size_t)(m0 + arow) * K + aseg;

  // prologue: stage k-block 0
  async_copy16(arp, &As[0][arow][aseg]);
  async_copy16(arp + 8, &As[0][arow][aseg + 8]);
  {
    const __bf16* bp = B + (size_t)brow * N + n0 + bseg;
    bf16x8 b0 = *(const bf16x8*)bp;
    bf16x8 b1 = *(const bf16x8*)(bp + 8);
#pragma unroll
    for (int j = 0; j < 8; ++j) Bs[0][bseg + j][brow] = b0[j];
#pragma unroll
    for (int j = 0; j < 8; ++j) Bs[0][bseg + 8 + j][brow] = b1[j];
  }
  wait_async0();
  __syncthreads();

  const int nk = K >> 5;
  for (int ks = 0; ks < nk; ++ks) {
    const int cur = ks & 1, nxt = cur ^ 1;
    const bool more = (ks + 1 < nk);
    bf16x8 nb0, nb1;
    if (more) {
      const int kk2 = (ks + 1) * 32;
      async_copy16(arp + kk2, &As[nxt][arow][aseg]);
      async_copy16(arp + kk2 + 8, &As[nxt][arow][aseg + 8]);
      const __bf16* bp = B + (size_t)(kk2 + brow) * N + n0 + bseg;
      nb0 = *(const bf16x8*)bp;
      nb1 = *(const bf16x8*)(bp + 8);
    }

    bf16x16 af = frag16(&As[cur][w * 16 + ln][kb]);
#pragma unroll
    for (int nt = 0; nt < 8; ++nt) {
      bf16x16 bfr = frag16(&Bs[cur][nt * 16 + ln][kb]);
      acc[nt] = wmma_bf16(af, bfr, acc[nt]);
    }

    if (more) {
#pragma unroll
      for (int j = 0; j < 8; ++j) Bs[nxt][bseg + j][brow] = nb0[j];
#pragma unroll
      for (int j = 0; j < 8; ++j) Bs[nxt][bseg + 8 + j][brow] = nb1[j];
    }
    wait_async0();
    __syncthreads();
  }

  const int gr = m0 + w * 16;
#pragma unroll
  for (int nt = 0; nt < 8; ++nt) {
#pragma unroll
    for (int r = 0; r < 8; ++r) {
      const int row = gr + r + 8 * hs;
      const int col = n0 + nt * 16 + ln;
      const float v = acc[nt][r];
      if constexpr (EPI == 0) {
        const int b = row >> 11, t = row & 2047;
        if (col < 1024) {
          if (t >= 1024) {
            const size_t o = (((size_t)b * 1024) + (t - 1024)) * 1024 + col;
            o0[o] = (__bf16)(v + bias0[col]);
            o1[o] = (__bf16)(v + bias1[col]);
          }
        } else if (col < 2048) {
          o2[(((size_t)b * 2048) + t) * 1024 + (col - 1024)] = (__bf16)v;
        } else {
          o3[(((size_t)b * 2048) + t) * 1024 + (col - 2048)] = (__bf16)v;
        }
      } else if constexpr (EPI == 1) {
        o0[(size_t)row * N + col] = (__bf16)v;
      } else {
        of[(size_t)row * N + col] = v;
      }
    }
  }
}

// ---------------------------------------------------------------- fused rel-attention
// Grid (8 qtiles, 16 heads, 4 batch), 256 threads = 8 waves; wave w owns q rows i0..i0+15.
// Transposed-S flash attention; rel-shift via 48-row Toeplitz band of rk.
// K chunk double-buffered via async global->LDS; V^T double-buffered via regs.
__global__ __launch_bounds__(256) void attn_kernel(
    const __bf16* __restrict__ qrw, const __bf16* __restrict__ qrr,
    const __bf16* __restrict__ kbf, const __bf16* __restrict__ vbf,
    const __bf16* __restrict__ rk, __bf16* __restrict__ av) {
  __shared__ __align__(16) __bf16 Ks[2][32][72];    // k chunk: [key][d]
  __shared__ __align__(16) __bf16 VTs[2][64][40];   // v^T chunk: [d][key]
  __shared__ float Band[8][48][19];                 // per-wave bd band tiles (C-layout)

  const int b = blockIdx.z, h = blockIdx.y;
  const int Q0 = blockIdx.x * 128;
  const int tid = threadIdx.x;
  const int w = tid >> 5, lane = tid & 31;
  const int hs = lane >> 4, ln = lane & 15, kb = hs * 8;
  const int i0 = Q0 + w * 16;

  // Persistent B fragments of (q + r_w_bias) and (q + r_r_bias): lane = q column.
  const __bf16* qr_rw = qrw + (((size_t)b * 1024 + i0 + ln) * 1024 + (size_t)h * 64);
  const __bf16* qr_rr = qrr + (((size_t)b * 1024 + i0 + ln) * 1024 + (size_t)h * 64);
  bf16x16 Brw[2], Brr[2];
#pragma unroll
  for (int kc = 0; kc < 2; ++kc) {
    Brw[kc] = frag16(qr_rw + 32 * kc + kb);
    Brr[kc] = frag16(qr_rr + 32 * kc + kb);
  }

  f32x8 Oz = {0.f, 0.f, 0.f, 0.f, 0.f, 0.f, 0.f, 0.f};
  f32x8 O[4] = {Oz, Oz, Oz, Oz};   // O^T d-tiles: lane = q, slot = d
  float m = -1e30f, l = 0.f;

  const size_t kvbase = ((size_t)b * 2048) * 1024 + (size_t)h * 64;
  const __bf16* rkh = rk + (size_t)h * 64;
  const int nch = (Q0 + 1152) >> 5;   // causal frontier: max valid key = 1024+Q0+127

  const int ldr = tid >> 3;           // cooperative kv load: key row 0..31
  const int lseg = (tid & 7) * 8;     // d segment
  const __bf16* kldp = kbf + kvbase + (size_t)ldr * 1024 + lseg;
  const __bf16* vldp = vbf + kvbase + (size_t)ldr * 1024 + lseg;

  // prologue: stage chunk 0
  async_copy16(kldp, &Ks[0][ldr][lseg]);
  {
    bf16x8 vd = *(const bf16x8*)vldp;
#pragma unroll
    for (int j = 0; j < 8; ++j) VTs[0][lseg + j][ldr] = vd[j];
  }
  wait_async0();
  __syncthreads();

  for (int ch = 0; ch < nch; ++ch) {
    const int kk = ch * 32;
    const int cur = ch & 1, nxt = cur ^ 1;
    const bool more = (ch + 1 < nch);
    bf16x8 vnext;
    if (more) {
      async_copy16(kldp + (size_t)(kk + 32) * 1024, &Ks[nxt][ldr][lseg]);
      vnext = *(const bf16x8*)(vldp + (size_t)(kk + 32) * 1024);
    }

    // ac^T = K_chunk(32x64) @ Qrw^T(64x16): lanes = q, slots = key
    f32x8 Sa = Oz, Sb = Oz;
#pragma unroll
    for (int kc = 0; kc < 2; ++kc) {
      Sa = wmma_bf16(frag16(&Ks[cur][ln][32 * kc + kb]), Brw[kc], Sa);
      Sb = wmma_bf16(frag16(&Ks[cur][16 + ln][32 * kc + kb]), Brw[kc], Sb);
    }

    // bd band: T^T[c,i] = rk_c . qrr_i  for c in [cb, cb+47]  (cb >= 0 by construction)
    const int cb = 1008 + kk - i0;
#pragma unroll
    for (int u = 0; u < 3; ++u) {
      int crow = cb + 16 * u + ln;
      crow = crow > 2047 ? 2047 : crow;   // over-read only in masked region
      const __bf16* rp = rkh + (size_t)crow * 1024;
      f32x8 T = Oz;
#pragma unroll
      for (int kc = 0; kc < 2; ++kc) T = wmma_bf16(frag16(rp + 32 * kc + kb), Brr[kc], T);
#pragma unroll
      for (int r = 0; r < 8; ++r) Band[w][16 * u + r + 8 * hs][ln] = T[r];
    }
    asm volatile("s_wait_dscnt 0x0" ::: "memory");   // same-wave LDS RAW

    // gather diagonal c = 1023 + j - i, mask j - i <= m_len, scale by 1/sqrt(64)
    float Sv[16];
#pragma unroll
    for (int t = 0; t < 2; ++t) {
#pragma unroll
      for (int r = 0; r < 8; ++r) {
        const int jrel = 16 * t + r + 8 * hs;
        const float bd = Band[w][15 + jrel - ln][ln];
        const float ac = t ? Sb[r] : Sa[r];
        const int jmi = (kk + jrel) - (i0 + ln);
        Sv[t * 8 + r] = (jmi <= 1024) ? (ac + bd) * 0.125f : -1e30f;
      }
    }

    // online softmax: per-lane stats (lane = q), merge half-waves with one shfl_xor
    float mc = Sv[0];
#pragma unroll
    for (int e = 1; e < 16; ++e) mc = fmaxf(mc, Sv[e]);
    mc = fmaxf(mc, __shfl_xor(mc, 16, 32));
    const float mn = fmaxf(m, mc);
    const float alpha = __expf(m - mn);
    float ps[16], lsum = 0.f;
#pragma unroll
    for (int e = 0; e < 16; ++e) {
      ps[e] = __expf(Sv[e] - mn);
      lsum += ps[e];
    }
    lsum += __shfl_xor(lsum, 16, 32);
    l = l * alpha + lsum;
    m = mn;
#pragma unroll
    for (int dt = 0; dt < 4; ++dt)
#pragma unroll
      for (int r = 0; r < 8; ++r) O[dt][r] *= alpha;

    // P^T in C-layout == B-operand layout: slot r -> B element r (tile a) / 8+r (tile b)
    bf16x16 Pb;
#pragma unroll
    for (int e = 0; e < 16; ++e) Pb[e] = (__bf16)ps[e];

    // O^T += V^T(64x32) @ P^T(32x16)
#pragma unroll
    for (int dt = 0; dt < 4; ++dt)
      O[dt] = wmma_bf16(frag16(&VTs[cur][dt * 16 + ln][kb]), Pb, O[dt]);

    if (more) {
#pragma unroll
      for (int j = 0; j < 8; ++j) VTs[nxt][lseg + j][ldr] = vnext[j];
    }
    wait_async0();
    __syncthreads();
  }

  const float linv = 1.f / l;
  const size_t ob = ((size_t)b * 1024 + i0 + ln) * 1024 + (size_t)h * 64;
#pragma unroll
  for (int dt = 0; dt < 4; ++dt)
#pragma unroll
    for (int r = 0; r < 8; ++r)
      av[ob + dt * 16 + r + 8 * hs] = (__bf16)(O[dt][r] * linv);
}

// ---------------------------------------------------------------- launch
extern "C" void kernel_launch(void* const* d_in, const int* in_sizes, int n_in,
                              void* d_out, int out_size, void* d_ws, size_t ws_size,
                              hipStream_t stream) {
  const float* inputs  = (const float*)d_in[0];   // (4,1024,1024)
  const float* pos_emb = (const float*)d_in[1];   // (1,2048,1024)
  const float* rwb     = (const float*)d_in[2];   // (16,64)
  const float* rrb     = (const float*)d_in[3];   // (16,64)
  const float* mems    = (const float*)d_in[4];   // (4,1024,1024)
  const float* W_w     = (const float*)d_in[5];   // (1024,3072)
  const float* W_r     = (const float*)d_in[6];   // (1024,1024)
  const float* W_o     = (const float*)d_in[7];   // (1024,1024)

  char* p = (char*)d_ws;
  auto take = [&](size_t bytes) { char* r = p; p += bytes; return r; };
  __bf16* cat_bf = (__bf16*)take(8192ull * 1024 * 2);   // (B*2048, 1024)
  __bf16* Ww_bf  = (__bf16*)take(1024ull * 3072 * 2);
  __bf16* pe_bf  = (__bf16*)take(2048ull * 1024 * 2);
  __bf16* Wr_bf  = (__bf16*)take(1024ull * 1024 * 2);
  __bf16* Wo_bf  = (__bf16*)take(1024ull * 1024 * 2);
  __bf16* qrw_bf = (__bf16*)take(4096ull * 1024 * 2);   // (B*1024, H*64) q + r_w_bias
  __bf16* qrr_bf = (__bf16*)take(4096ull * 1024 * 2);   // q + r_r_bias
  __bf16* k_bf   = (__bf16*)take(8192ull * 1024 * 2);   // (B*2048, H*64)
  __bf16* v_bf   = (__bf16*)take(8192ull * 1024 * 2);
  __bf16* rk_bf  = (__bf16*)take(2048ull * 1024 * 2);   // (2048, H*64)
  __bf16* av_bf  = (__bf16*)take(4096ull * 1024 * 2);   // attn_vec (B*1024, 1024)

  // pack to bf16 (x8 vectorized)
  build_cat<<<8388608 / (256 * 8), 256, 0, stream>>>(mems, inputs, cat_bf);
  cvt_f32_bf16<<<3145728 / (256 * 8), 256, 0, stream>>>(W_w, Ww_bf, 3145728);
  cvt_f32_bf16<<<2097152 / (256 * 8), 256, 0, stream>>>(pos_emb, pe_bf, 2097152);
  cvt_f32_bf16<<<1048576 / (256 * 8), 256, 0, stream>>>(W_r, Wr_bf, 1048576);
  cvt_f32_bf16<<<1048576 / (256 * 8), 256, 0, stream>>>(W_o, Wo_bf, 1048576);

  // w_heads = cat @ W_w, scattered into q(+biases)/k/v
  gemm_bf16_kernel<0><<<dim3(3072 / 128, 8192 / 128), 256, 0, stream>>>(
      cat_bf, Ww_bf, 8192, 3072, 1024, qrw_bf, qrr_bf, k_bf, v_bf, nullptr, rwb, rrb);

  // r_head_k = pe @ W_r
  gemm_bf16_kernel<1><<<dim3(1024 / 128, 2048 / 128), 256, 0, stream>>>(
      pe_bf, Wr_bf, 2048, 1024, 1024, rk_bf, nullptr, nullptr, nullptr, nullptr,
      nullptr, nullptr);

  // fused rel-shift masked softmax attention
  attn_kernel<<<dim3(8, 16, 4), 256, 0, stream>>>(qrw_bf, qrr_bf, k_bf, v_bf, rk_bf, av_bf);

  // out = attn_vec @ W_o  (f32)
  gemm_bf16_kernel<2><<<dim3(1024 / 128, 4096 / 128), 256, 0, stream>>>(
      av_bf, Wo_bf, 4096, 1024, 1024, nullptr, nullptr, nullptr, nullptr,
      (float*)d_out, nullptr, nullptr);

  (void)in_sizes; (void)n_in; (void)out_size; (void)ws_size;
}
